// MultiModalLIMoE_1013612282519
// MI455X (gfx1250) — matrix operand order repaired and checked by reading, not compiled, not run on hardware
//
#include <hip/hip_runtime.h>

#define TOK 8192
#define DM  512
#define SEQ 1024
#define LSK 40   // LDS row stride (bf16 elements): 16B-aligned rows, conflict-free b128 access

typedef __bf16 bf16_t;
typedef bf16_t bf16x16 __attribute__((ext_vector_type(16)));
typedef bf16_t bf16x2  __attribute__((ext_vector_type(2)));
typedef float  f32x2   __attribute__((ext_vector_type(2)));
typedef float  v8f     __attribute__((ext_vector_type(8)));

__device__ __forceinline__ bf16_t f2bf(float f) { return (bf16_t)f; }

// packed f32x2 -> bf16x2 (lowers to v_cvt_pk_bf16_f32)
__device__ __forceinline__ bf16x2 cvt2(float a, float b) {
  f32x2 f; f.x = a; f.y = b;
  return __builtin_convertvector(f, bf16x2);
}

// convert 16 floats (4 x float4) -> 16 bf16 packed in two uint4
union Pack16 { bf16x2 h2[8]; uint4 q[2]; };
__device__ __forceinline__ void pack16(Pack16& u, const float4* v) {
#pragma unroll
  for (int i = 0; i < 4; ++i) {
    u.h2[2 * i]     = cvt2(v[i].x, v[i].y);
    u.h2[2 * i + 1] = cvt2(v[i].z, v[i].w);
  }
}

__device__ __forceinline__ bf16x16 ldfrag(const bf16_t* p0, const bf16_t* p1) {
  union { uint4 q[2]; bf16x16 v; } u;
  u.q[0] = *(const uint4*)p0;
  u.q[1] = *(const uint4*)p1;
  return u.v;
}

__device__ __forceinline__ v8f wmma_bf16(bf16x16 a, bf16x16 b, v8f c) {
  return __builtin_amdgcn_wmma_f32_16x16x32_bf16(false, a, false, b, (short)0, c, false, false);
}

// ---------------------------------------------------------------------------
// Weight convert+transpose: Wt[n][k] (bf16) = W[k][n] (f32).
// grid (N/32, K/32, Z); Z indexes layer/expert slabs via strides.
// ---------------------------------------------------------------------------
__global__ __launch_bounds__(256) void convt_k(const float* __restrict__ W, bf16_t* __restrict__ Wt,
                                               int K, int N, long sin, long sout)
{
  __shared__ float tile[32][33];
  const float* We = W + (long)blockIdx.z * sin;
  bf16_t* Wte = Wt + (long)blockIdx.z * sout;
  int n0 = blockIdx.x * 32, k0 = blockIdx.y * 32;
  int tx = threadIdx.x & 31;
  int ty = threadIdx.x >> 5;
#pragma unroll
  for (int i = 0; i < 4; ++i)
    tile[ty + 8 * i][tx] = We[(long)(k0 + ty + 8 * i) * N + n0 + tx];
  __syncthreads();
#pragma unroll
  for (int i = 0; i < 4; ++i)
    Wte[(long)(n0 + ty + 8 * i) * K + k0 + tx] = f2bf(tile[tx][ty + 8 * i]);
}

// ---------------------------------------------------------------------------
// Tiled GEMM: C[M,N] = act(A[M,K] @ W[K,N] + bias), W given pre-transposed
// bf16 Wt[N][K]. Optional per-expert row gather (top-1 MoE).
// Block 256 threads (8 waves), tile 128x128, K-step 32, LDS double-buffered.
// Gather rows beyond cnt[e] are clamped to a valid row (their results are
// discarded in the epilogue; WMMA rows are independent, so no zero-fill).
// ---------------------------------------------------------------------------
__global__ __launch_bounds__(256) void gemm_k(
    const float* __restrict__ A, const bf16_t* __restrict__ Wt,
    const float* __restrict__ bias, float* __restrict__ C,
    int M, int N, int K, int lda, int ldc, int relu,
    const int* __restrict__ bucket, const int* __restrict__ cnt,
    long wstride, long bstride)
{
  __shared__ bf16_t As[2][128 * LSK];
  __shared__ bf16_t Bs[2][128 * LSK];

  const int e = blockIdx.z;
  const bf16_t* We = Wt + (long)e * wstride;
  const float* be = bias ? (bias + (long)e * bstride) : nullptr;
  const int* buck = bucket ? (bucket + e * M) : nullptr;
  const int rcnt = buck ? cnt[e] : M;

  const int m0 = blockIdx.x * 128;
  const int n0 = blockIdx.y * 128;
  if (m0 >= rcnt) return;                 // uniform per block; implies rcnt >= 1 below

  const int tid   = threadIdx.x;
  const int lane  = tid & 31;
  const int wave  = tid >> 5;
  const int lrow  = lane & 15;
  const int lhalf = lane >> 4;
  const int wm = (wave & 3) * 32;         // wave row base in tile
  const int wn = (wave >> 2) * 64;        // wave col base in tile

  // A-tile load mapping: row = tid&127, 16 cols per half; clamped gather
  const int ar = tid & 127;
  const int ac = (tid >> 7) * 16;
  const float* aptr;
  {
    int gr = m0 + ar;
    int gc = gr < rcnt ? gr : rcnt - 1;   // safe row; result unused for gr >= rcnt
    int tok = buck ? buck[gc] : gc;
    aptr = A + (long)tok * lda;
  }
  // B-tile load mapping: n = tid>>1, 16-element K half per thread (pure copy)
  const int bn = tid >> 1;
  const int bk = (tid & 1) * 16;

  auto loadA = [&](int buf, int k0) {
    float4 v[4];
#pragma unroll
    for (int i = 0; i < 4; ++i) v[i] = *(const float4*)(aptr + k0 + ac + i * 4);
    Pack16 u;
    pack16(u, v);
    *(uint4*)&As[buf][ar * LSK + ac]     = u.q[0];
    *(uint4*)&As[buf][ar * LSK + ac + 8] = u.q[1];
  };
  auto loadB = [&](int buf, int k0) {
    const bf16_t* src = We + (long)(n0 + bn) * K + k0 + bk;
    uint4 u0 = *(const uint4*)src;
    uint4 u1 = *(const uint4*)(src + 8);
    *(uint4*)&Bs[buf][bn * LSK + bk]     = u0;
    *(uint4*)&Bs[buf][bn * LSK + bk + 8] = u1;
  };

  v8f acc[2][4];
#pragma unroll
  for (int i = 0; i < 2; ++i)
#pragma unroll
    for (int j = 0; j < 4; ++j)
#pragma unroll
      for (int r = 0; r < 8; ++r) acc[i][j][r] = 0.f;

  loadA(0, 0);
  loadB(0, 0);
  __syncthreads();

  int cur = 0;
  for (int k0 = 0; k0 < K; k0 += 32) {
    if (k0 + 32 < K) {                    // prefetch next tile into other buffer
      loadA(cur ^ 1, k0 + 32);
      loadB(cur ^ 1, k0 + 32);
    }
    bf16x16 af[2], bfr[4];
#pragma unroll
    for (int i = 0; i < 2; ++i) {
      const bf16_t* p = &As[cur][(wm + i * 16 + lrow) * LSK + lhalf * 16];
      af[i] = ldfrag(p, p + 8);           // A-layout: K contiguous per half {16h..16h+15}
    }
#pragma unroll
    for (int j = 0; j < 4; ++j) {
      const bf16_t* p = &Bs[cur][(wn + j * 16 + lrow) * LSK + lhalf * 16];
      bfr[j] = ldfrag(p, p + 8);          // B-layout: lane = col, K contiguous per half (same order as A)
    }
#pragma unroll
    for (int i = 0; i < 2; ++i)
#pragma unroll
      for (int j = 0; j < 4; ++j)
        acc[i][j] = wmma_bf16(af[i], bfr[j], acc[i][j]);
    __syncthreads();
    cur ^= 1;
  }

  // Epilogue: C/D layout M = r + 8*(lane/16), N = lane%16 (per 16x16 tile)
#pragma unroll
  for (int j = 0; j < 4; ++j) {
    int n = n0 + wn + j * 16 + lrow;
    float bv = be ? be[n] : 0.f;
#pragma unroll
    for (int i = 0; i < 2; ++i) {
#pragma unroll
      for (int r = 0; r < 8; ++r) {
        int mt = m0 + wm + i * 16 + r + 8 * lhalf;
        if (mt >= rcnt) continue;
        int tok = buck ? buck[mt] : mt;
        float v = acc[i][j][r] + bv;
        if (relu) v = fmaxf(v, 0.f);
        C[(long)tok * ldc + n] = v;
      }
    }
  }
}

// ---------------------------------------------------------------------------
// Flash attention: grid (S/64, H, B), 128 threads = 4 waves, 16 queries/wave.
// QKV layout: token-major [8192][1536], q|k|v each 512 wide, head h at h*64.
// ---------------------------------------------------------------------------
__global__ __launch_bounds__(128) void attn_k(const float* __restrict__ QKV,
                                              float* __restrict__ CTX)
{
  __shared__ bf16_t Qs[64 * 72];           // [qrow][hd]
  __shared__ bf16_t Ks[32 * 72];           // [key][hd]
  __shared__ bf16_t Vt[64 * 40];           // [hd][key]
  __shared__ float SP[4][16][36];          // per-wave score staging
  __shared__ bf16_t Pb[4][16 * 40];        // per-wave probs (A-layout source)
  __shared__ float alpha_s[4][16];
  __shared__ float linv_s[4][16];

  const int q0 = blockIdx.x * 64;
  const int h  = blockIdx.y;
  const int b  = blockIdx.z;
  const float* base = QKV + (long)b * SEQ * 1536;
  const int tid   = threadIdx.x;
  const int wave  = tid >> 5;
  const int lane  = tid & 31;
  const int lrow  = lane & 15;
  const int lhalf = lane >> 4;

  { // load Q block (64x64) -> bf16 LDS, packed converts + b128 stores
    int r = tid & 63;
    int cb = (tid >> 6) * 32;
    const float* qp = base + (long)(q0 + r) * 1536 + h * 64 + cb;
#pragma unroll
    for (int g = 0; g < 2; ++g) {
      float4 v[4];
#pragma unroll
      for (int i = 0; i < 4; ++i) v[i] = *(const float4*)(qp + g * 16 + i * 4);
      Pack16 u;
      pack16(u, v);
      *(uint4*)&Qs[r * 72 + cb + g * 16]     = u.q[0];
      *(uint4*)&Qs[r * 72 + cb + g * 16 + 8] = u.q[1];
    }
  }

  v8f acc[4];
#pragma unroll
  for (int j = 0; j < 4; ++j)
#pragma unroll
    for (int r = 0; r < 8; ++r) acc[j][r] = 0.f;
  float mrow = -3.0e38f, lsum = 0.f;       // live in lanes 0..15

  for (int t0 = 0; t0 < SEQ; t0 += 32) {
    __syncthreads();                       // protect K/V tiles from prior reads
    {
      int key = tid & 31;
      int cb = (tid >> 5) * 16;
      const float* kp = base + (long)(t0 + key) * 1536 + 512  + h * 64 + cb;
      const float* vp = base + (long)(t0 + key) * 1536 + 1024 + h * 64 + cb;
      float4 v[4];
#pragma unroll
      for (int i = 0; i < 4; ++i) v[i] = *(const float4*)(kp + i * 4);
      Pack16 u;
      pack16(u, v);
      *(uint4*)&Ks[key * 72 + cb]     = u.q[0];
      *(uint4*)&Ks[key * 72 + cb + 8] = u.q[1];
#pragma unroll
      for (int i = 0; i < 4; ++i) {        // V transpose: scattered b16 stores
        float4 w = *(const float4*)(vp + i * 4);
        Vt[(cb + i * 4 + 0) * 40 + key] = f2bf(w.x);
        Vt[(cb + i * 4 + 1) * 40 + key] = f2bf(w.y);
        Vt[(cb + i * 4 + 2) * 40 + key] = f2bf(w.z);
        Vt[(cb + i * 4 + 3) * 40 + key] = f2bf(w.w);
      }
    }
    __syncthreads();

    // scores = Q(16x64) @ K^T(64x32): 4 wmma
    v8f s0, s1;
#pragma unroll
    for (int r = 0; r < 8; ++r) { s0[r] = 0.f; s1[r] = 0.f; }
#pragma unroll
    for (int kc = 0; kc < 2; ++kc) {
      const bf16_t* qp = &Qs[(wave * 16 + lrow) * 72 + kc * 32 + lhalf * 16];
      bf16x16 aq = ldfrag(qp, qp + 8);
      const bf16_t* k0p = &Ks[lrow * 72 + kc * 32 + lhalf * 16];
      const bf16_t* k1p = &Ks[(16 + lrow) * 72 + kc * 32 + lhalf * 16];
      s0 = wmma_bf16(aq, ldfrag(k0p, k0p + 8), s0);
      s1 = wmma_bf16(aq, ldfrag(k1p, k1p + 8), s1);
    }
#pragma unroll
    for (int r = 0; r < 8; ++r) {
      SP[wave][r + 8 * lhalf][lrow]      = s0[r] * 0.125f;   // 1/sqrt(64)
      SP[wave][r + 8 * lhalf][16 + lrow] = s1[r] * 0.125f;
    }
    __syncthreads();

    // online softmax (one lane per query row)
    if (lhalf == 0) {
      float mx = mrow;
#pragma unroll
      for (int c = 0; c < 32; ++c) mx = fmaxf(mx, SP[wave][lrow][c]);
      float p[32];
      float sum = 0.f;
#pragma unroll
      for (int c = 0; c < 32; ++c) {
        p[c] = __expf(SP[wave][lrow][c] - mx);
        sum += p[c];
      }
      union { bf16x2 h2[16]; uint4 q[4]; } up;
#pragma unroll
      for (int c2 = 0; c2 < 16; ++c2) up.h2[c2] = cvt2(p[2 * c2], p[2 * c2 + 1]);
#pragma unroll
      for (int qi = 0; qi < 4; ++qi) *(uint4*)&Pb[wave][lrow * 40 + qi * 8] = up.q[qi];
      float alpha = __expf(mrow - mx);
      lsum = lsum * alpha + sum;
      mrow = mx;
      alpha_s[wave][lrow] = alpha;
    }
    __syncthreads();

    // rescale running context, then ctx += P(16x32) @ V(32x64): 4 wmma
#pragma unroll
    for (int r = 0; r < 8; ++r) {
      float al = alpha_s[wave][r + 8 * lhalf];
#pragma unroll
      for (int j = 0; j < 4; ++j) acc[j][r] *= al;
    }
    const bf16_t* pp = &Pb[wave][lrow * 40 + lhalf * 16];
    bf16x16 pf = ldfrag(pp, pp + 8);
#pragma unroll
    for (int j = 0; j < 4; ++j) {
      const bf16_t* vpr = &Vt[(j * 16 + lrow) * 40 + lhalf * 16];
      acc[j] = wmma_bf16(pf, ldfrag(vpr, vpr + 8), acc[j]);
    }
  }

  if (lhalf == 0) linv_s[wave][lrow] = 1.f / lsum;
  __syncthreads();
#pragma unroll
  for (int r = 0; r < 8; ++r) {
    int rr = r + 8 * lhalf;
    float inv = linv_s[wave][rr];
    int row = q0 + wave * 16 + rr;
#pragma unroll
    for (int j = 0; j < 4; ++j)
      CTX[((long)(b * SEQ + row)) * DM + h * 64 + j * 16 + lrow] = acc[j][r] * inv;
  }
}

// ---------------------------------------------------------------------------
// X = LN(X + R) * g + beta ; one wave per 512-wide row, 8 rows per block.
// ---------------------------------------------------------------------------
__global__ __launch_bounds__(256) void ln_k(float* __restrict__ X, const float* __restrict__ R,
                                            const float* __restrict__ g, const float* __restrict__ beta)
{
  int row = blockIdx.x * 8 + (threadIdx.x >> 5);
  int lane = threadIdx.x & 31;
  float* xr = X + (long)row * DM;
  const float* rr = R + (long)row * DM;
  float v[16];
  float s = 0.f;
#pragma unroll
  for (int i = 0; i < 16; ++i) { int c = lane + i * 32; v[i] = xr[c] + rr[c]; s += v[i]; }
#pragma unroll
  for (int o = 16; o > 0; o >>= 1) s += __shfl_xor(s, o, 32);
  float mu = s * (1.f / 512.f);
  float var = 0.f;
#pragma unroll
  for (int i = 0; i < 16; ++i) { float d = v[i] - mu; var += d * d; }
#pragma unroll
  for (int o = 16; o > 0; o >>= 1) var += __shfl_xor(var, o, 32);
  float rinv = rsqrtf(var * (1.f / 512.f) + 1e-5f);
#pragma unroll
  for (int i = 0; i < 16; ++i) { int c = lane + i * 32; xr[c] = (v[i] - mu) * rinv * g[c] + beta[c]; }
}

// Gate logits + argmax (softmax is monotonic) + atomic bucketing by expert.
__global__ __launch_bounds__(256) void gate_k(const float* __restrict__ X, const float* __restrict__ gw,
                                              const float* __restrict__ gb, int* __restrict__ cnt,
                                              int* __restrict__ bucket)
{
  int t = blockIdx.x * 256 + threadIdx.x;
  const float* x = X + (long)t * DM;
  float a[8];
#pragma unroll
  for (int e = 0; e < 8; ++e) a[e] = gb[e];
  for (int k = 0; k < DM; ++k) {
    float xv = x[k];
    const float* w = gw + k * 8;
#pragma unroll
    for (int e = 0; e < 8; ++e) a[e] += xv * w[e];
  }
  int best = 0; float bv = a[0];
#pragma unroll
  for (int e = 1; e < 8; ++e) if (a[e] > bv) { bv = a[e]; best = e; }
  int pos = atomicAdd(&cnt[best], 1);
  bucket[best * TOK + pos] = t;
}

__global__ void zero8_k(int* cnt) { if (threadIdx.x < 8) cnt[threadIdx.x] = 0; }

__global__ __launch_bounds__(256) void embed_k(const float* __restrict__ modp, const float* __restrict__ ew,
                                               const float* __restrict__ eb, float* __restrict__ X)
{
  long idx = (long)blockIdx.x * 256 + threadIdx.x;  // over 8192*512
  int d = (int)(idx & 511);
  int t = (int)(idx >> 9);
  int b = t >> 10, p = t & 1023;
  float s = eb[d];
#pragma unroll
  for (int c = 0; c < 3; ++c) s += modp[((long)(b * 3 + c)) * 1024 + p] * ew[c * DM + d];
  X[idx] = s;
}

__global__ __launch_bounds__(512) void mean_k(const float* __restrict__ X, float* __restrict__ repr)
{
  int b = blockIdx.x, d = threadIdx.x;
  float s = 0.f;
  for (int t = 0; t < SEQ; ++t) s += X[((long)(b * SEQ + t)) * DM + d];
  repr[b * DM + d] = s * (1.f / 1024.f);
}

__global__ __launch_bounds__(512) void cls1_k(const float* __restrict__ reprs, const float* __restrict__ w1,
                                              const float* __restrict__ b1, float* __restrict__ hb)
{
  int b = blockIdx.x, n = threadIdx.x;
  float s = b1[n];
  for (int j = 0; j < 512; ++j) s += reprs[b * 512 + j] * w1[j * 512 + n];
  for (int j = 0; j < 512; ++j) s += reprs[4096 + b * 512 + j] * w1[(512 + j) * 512 + n];
  hb[b * 512 + n] = fmaxf(s, 0.f);
}

__global__ __launch_bounds__(256) void cls2_k(const float* __restrict__ hb, const float* __restrict__ w2,
                                              const float* __restrict__ b2, float* __restrict__ out)
{
  int b = blockIdx.x;
  int n = blockIdx.y * 256 + threadIdx.x;
  if (n >= 1000) return;
  float s = b2[n];
  for (int j = 0; j < 512; ++j) s += hb[b * 512 + j] * w2[j * 1000 + n];
  out[b * 1000 + n] = s;
}

extern "C" void kernel_launch(void* const* d_in, const int* in_sizes, int n_in,
                              void* d_out, int out_size, void* d_ws, size_t ws_size,
                              hipStream_t stream)
{
  (void)in_sizes; (void)n_in; (void)out_size; (void)ws_size;
  const float* mods[2] = {(const float*)d_in[0], (const float*)d_in[1]};
  const float* embed_w = (const float*)d_in[2];
  const float* embed_b = (const float*)d_in[3];
  const float* qkv_w   = (const float*)d_in[4];
  const float* qkv_b   = (const float*)d_in[5];
  const float* out_w   = (const float*)d_in[6];
  const float* out_b   = (const float*)d_in[7];
  const float* lin1_w  = (const float*)d_in[8];
  const float* lin1_b  = (const float*)d_in[9];
  const float* lin2_w  = (const float*)d_in[10];
  const float* lin2_b  = (const float*)d_in[11];
  const float* ln1_g   = (const float*)d_in[12];
  const float* ln1_bt  = (const float*)d_in[13];
  const float* ln2_g   = (const float*)d_in[14];
  const float* ln2_bt  = (const float*)d_in[15];
  const float* gate_w  = (const float*)d_in[16];
  const float* gate_b  = (const float*)d_in[17];
  const float* moe_w1  = (const float*)d_in[18];
  const float* moe_b1  = (const float*)d_in[19];
  const float* moe_w2  = (const float*)d_in[20];
  const float* moe_b2  = (const float*)d_in[21];
  const float* cls_w1  = (const float*)d_in[22];
  const float* cls_b1  = (const float*)d_in[23];
  const float* cls_w2  = (const float*)d_in[24];
  const float* cls_b2  = (const float*)d_in[25];

  // workspace layout (~181 MB): f32 activations + bf16 transposed weights
  float* X    = (float*)d_ws;                  // [8192][512]
  float* Y    = X + (size_t)TOK * DM;          // [8192][2048] (QKV / FFN hidden / MoE hidden)
  float* CTX  = Y + (size_t)TOK * 1536;        // aliased into Y tail (disjoint from QKV cols)
  float* Z    = Y + (size_t)TOK * 2048;        // [8192][512]
  int*   cnt    = (int*)(Z + (size_t)TOK * DM);
  int*   bucket = cnt + 16;                    // [8][8192]
  float* reprs  = (float*)(bucket + 8 * TOK);  // [2][8][512]
  float* clsh   = reprs + 2 * 8 * DM;          // [8][512]
  bf16_t* qkvT  = (bf16_t*)(clsh + 8 * DM);    // [2][1536][512]
  bf16_t* outT  = qkvT  + (size_t)2 * 1536 * 512;          // [2][512][512]
  bf16_t* lin1T = outT  + (size_t)2 * 512 * 512;           // [2][2048][512]
  bf16_t* lin2T = lin1T + (size_t)2 * 2048 * 512;          // [2][512][2048]
  bf16_t* moe1T = lin2T + (size_t)2 * 512 * 2048;          // [2*8][2048][512]
  bf16_t* moe2T = moe1T + (size_t)16 * 2048 * 512;         // [2*8][512][2048]

  // One-shot weight convert+transpose to bf16 [N][K]
  convt_k<<<dim3(48, 16, 2),  256, 0, stream>>>(qkv_w,  qkvT,  512, 1536, (long)512 * 1536, (long)1536 * 512);
  convt_k<<<dim3(16, 16, 2),  256, 0, stream>>>(out_w,  outT,  512, 512,  (long)512 * 512,  (long)512 * 512);
  convt_k<<<dim3(64, 16, 2),  256, 0, stream>>>(lin1_w, lin1T, 512, 2048, (long)512 * 2048, (long)2048 * 512);
  convt_k<<<dim3(16, 64, 2),  256, 0, stream>>>(lin2_w, lin2T, 2048, 512, (long)2048 * 512, (long)512 * 2048);
  convt_k<<<dim3(64, 16, 16), 256, 0, stream>>>(moe_w1, moe1T, 512, 2048, (long)512 * 2048, (long)2048 * 512);
  convt_k<<<dim3(16, 64, 16), 256, 0, stream>>>(moe_w2, moe2T, 2048, 512, (long)2048 * 512, (long)512 * 2048);

  for (int m = 0; m < 2; ++m) {
    embed_k<<<TOK * DM / 256, 256, 0, stream>>>(mods[m], embed_w + m * 3 * DM, embed_b + m * DM, X);

    for (int l = 0; l < 2; ++l) {
      gemm_k<<<dim3(64, 12, 1), 256, 0, stream>>>(X, qkvT + (size_t)l * 1536 * 512, qkv_b + l * 1536,
                                                  Y, TOK, 1536, 512, 512, 1536, 0, nullptr, nullptr, 0, 0);
      attn_k<<<dim3(16, 8, 8), 128, 0, stream>>>(Y, CTX);
      gemm_k<<<dim3(64, 4, 1), 256, 0, stream>>>(CTX, outT + (size_t)l * 512 * 512, out_b + l * 512,
                                                 Z, TOK, 512, 512, 512, 512, 0, nullptr, nullptr, 0, 0);
      ln_k<<<1024, 256, 0, stream>>>(X, Z, ln1_g + l * 512, ln1_bt + l * 512);
      gemm_k<<<dim3(64, 16, 1), 256, 0, stream>>>(X, lin1T + (size_t)l * 2048 * 512, lin1_b + l * 2048,
                                                  Y, TOK, 2048, 512, 512, 2048, 1, nullptr, nullptr, 0, 0);
      gemm_k<<<dim3(64, 4, 1), 256, 0, stream>>>(Y, lin2T + (size_t)l * 512 * 2048, lin2_b + l * 512,
                                                 Z, TOK, 512, 2048, 2048, 512, 0, nullptr, nullptr, 0, 0);
      ln_k<<<1024, 256, 0, stream>>>(X, Z, ln2_g + l * 512, ln2_bt + l * 512);
    }

    for (int l = 0; l < 2; ++l) {
      zero8_k<<<1, 32, 0, stream>>>(cnt);
      gate_k<<<32, 256, 0, stream>>>(X, gate_w + (long)l * 512 * 8, gate_b + l * 8, cnt, bucket);
      // top-1 expert FFN (mathematically identical to dense-masked reference)
      gemm_k<<<dim3(64, 16, 8), 256, 0, stream>>>(X, moe1T + (size_t)l * 8 * 2048 * 512,
                                                  moe_b1 + (long)l * 8 * 2048, Y,
                                                  TOK, 2048, 512, 512, 2048, 1,
                                                  bucket, cnt, (long)2048 * 512, 2048);
      gemm_k<<<dim3(64, 4, 8), 256, 0, stream>>>(Y, moe2T + (size_t)l * 8 * 512 * 2048,
                                                 moe_b2 + (long)l * 8 * 512, X,
                                                 TOK, 512, 2048, 2048, 512, 0,
                                                 bucket, cnt, (long)512 * 2048, 512);
    }

    mean_k<<<8, 512, 0, stream>>>(X, reprs + m * 8 * DM);
  }

  cls1_k<<<8, 512, 0, stream>>>(reprs, cls_w1, cls_b1, clsh);
  cls2_k<<<dim3(8, 4), 256, 0, stream>>>(clsh, cls_w2, cls_b2, (float*)d_out);
}
